// SelfAttention_20040317403582
// MI455X (gfx1250) — matrix operand order repaired and checked by reading, not compiled, not run on hardware
//
#include <hip/hip_runtime.h>
#include <hip/hip_bf16.h>
#include <stdint.h>

typedef __attribute__((ext_vector_type(16))) __bf16 v16bf;
typedef __attribute__((ext_vector_type(8)))  float  v8f;

#define BB 8
#define NNTOK 2048
#define CCH 512

__device__ __forceinline__ __bf16 f32_to_bf16(float f) {
  union { float f; uint32_t u; } in{f};
  uint32_t u = in.u;
  u += 0x7FFFu + ((u >> 16) & 1u);     // round to nearest even
  union { uint16_t u; __bf16 b; } out{(uint16_t)(u >> 16)};
  return out.b;
}

union FragCvt {
  struct { uint4 lo; uint4 hi; } u;
  v16bf v;
};

// A-matrix 16x32 bf16 fragment (ISA 7.12.2): lane row = lane&15,
// phase h = lane>>4: elems 0..7 <- K = h*8..h*8+7, elems 8..15 <- K+16.
__device__ __forceinline__ v16bf load_frag_A(const __bf16* base) {
  FragCvt r;
  r.u.lo = *(const uint4*)(base);
  r.u.hi = *(const uint4*)(base + 16);
  return r.v;
}

// B-matrix 32x16 bf16 fragment: lane col = lane&15, elems 0..7 <- K = h*8..h*8+7, elems 8..15 <- K+16.
__device__ __forceinline__ v16bf load_frag_B(const __bf16* base) {
  FragCvt r;
  r.u.lo = *(const uint4*)(base);
  r.u.hi = *(const uint4*)(base + 16);
  return r.v;
}

// Generic -> LDS(AS3) byte offset for async-DMA VDST operands.
typedef __attribute__((address_space(3))) const void lds_cv_t;
__device__ __forceinline__ uint32_t lds_offset(const void* p) {
  return (uint32_t)(uintptr_t)(lds_cv_t*)p;
}

// ASYNCcnt-tracked global->LDS 128b copy (CDNA5 ISA).
__device__ __forceinline__ void async_copy_b128(uint32_t lds_addr, const void* gptr) {
  asm volatile("global_load_async_to_lds_b128 %0, %1, off"
               :: "v"(lds_addr), "v"((uint64_t)(uintptr_t)gptr)
               : "memory");
}
__device__ __forceinline__ void wait_asynccnt0() {
  asm volatile("s_wait_asynccnt 0x0" ::: "memory");
}

// ---------------------------------------------------------------------------
// fp32 -> bf16 conversion
// ---------------------------------------------------------------------------
__global__ void cvt_f32_bf16_kernel(const float* __restrict__ in,
                                    __bf16* __restrict__ out, long n) {
  long i = (long)blockIdx.x * blockDim.x + threadIdx.x;
  long stride = (long)gridDim.x * blockDim.x;
  for (; i < n; i += stride) out[i] = f32_to_bf16(in[i]);
}

// ---------------------------------------------------------------------------
// Register-blocked WMMA GEMM: out[g,m,n] = sum_k A[g,m,k] * Bm[g,n,k]
// 256 threads / 8 waves; block tile 256(m) x 128(n); wave tile 64x64
// (4x4 accumulators, 16 WMMA per K-step per wave); K step 32.
// Double-buffered LDS staged via global_load_async_to_lds_b128, so the
// next tile's DMA overlaps the current tile's WMMAs.
// ---------------------------------------------------------------------------
enum { MODE_BF16 = 0, MODE_RELU_BF16 = 1, MODE_F32 = 2, MODE_FINAL = 3,
       MODE_RESID = 4 };

template<int MODE>
__global__ __launch_bounds__(256)
void wmma_gemm_tiled(const __bf16* __restrict__ A, long sAg,
                     const __bf16* __restrict__ Bm, long sBg,
                     int M, int Nd, int K,
                     __bf16* __restrict__ outb, float* __restrict__ outf,
                     const float* __restrict__ res,
                     const float* __restrict__ scale) {
  __shared__ __align__(16) __bf16 lds_a[2][256][32];   // 2 x 16KB
  __shared__ __align__(16) __bf16 lds_b[2][128][32];   // 2 x  8KB

  const int g  = blockIdx.z;
  const int m0 = blockIdx.y * 256;
  const int n0 = blockIdx.x * 128;
  const int tid  = threadIdx.x;
  const int lane = tid & 31;
  const int wave = tid >> 5;
  const int wr = wave & 3;          // 4 wave-rows  -> 4 * 64 = 256 m
  const int wc = wave >> 2;         // 2 wave-cols  -> 2 * 64 = 128 n
  const int h  = lane >> 4;
  const int lo = lane & 15;

  const __bf16* Ag = A  + (long)g * sAg;
  const __bf16* Bg = Bm + (long)g * sBg;
  const uint32_t lA = lds_offset(&lds_a[0][0][0]);
  const uint32_t lB = lds_offset(&lds_b[0][0][0]);

  // Stage one 256x32 A tile (1024 x 16B) + one 128x32 B tile (512 x 16B)
  // into LDS buffer `buf`, 6 async b128 copies per thread.
  auto stage = [&](int buf, int kb) {
#pragma unroll
    for (int s = 0; s < 4; ++s) {                // A: chunks 0..1023
      const int chunk = s * 256 + tid;
      const int row = chunk >> 2, seg = chunk & 3;
      async_copy_b128(lA + (uint32_t)(buf * 16384 + row * 64 + seg * 16),
                      Ag + (long)(m0 + row) * K + kb + seg * 8);
    }
#pragma unroll
    for (int s = 0; s < 2; ++s) {                // B: chunks 0..511
      const int chunk = s * 256 + tid;
      const int row = chunk >> 2, seg = chunk & 3;
      async_copy_b128(lB + (uint32_t)(buf * 8192 + row * 64 + seg * 16),
                      Bg + (long)(n0 + row) * K + kb + seg * 8);
    }
  };

  v8f acc[4][4];
#pragma unroll
  for (int ms = 0; ms < 4; ++ms)
#pragma unroll
    for (int ns = 0; ns < 4; ++ns)
      acc[ms][ns] = (v8f){0.f, 0.f, 0.f, 0.f, 0.f, 0.f, 0.f, 0.f};

  stage(0, 0);
  wait_asynccnt0();
  __syncthreads();

  for (int kb = 0; kb < K; kb += 32) {
    const int cur = (kb >> 5) & 1;
    if (kb + 32 < K) stage(cur ^ 1, kb + 32);    // DMA next tile under math

    v16bf afrag[4], bfrag[4];
#pragma unroll
    for (int ms = 0; ms < 4; ++ms)
      afrag[ms] = load_frag_A(&lds_a[cur][wr * 64 + ms * 16 + lo][h * 8]);
#pragma unroll
    for (int ns = 0; ns < 4; ++ns)
      bfrag[ns] = load_frag_B(&lds_b[cur][wc * 64 + ns * 16 + lo][h * 8]);
#pragma unroll
    for (int ms = 0; ms < 4; ++ms)
#pragma unroll
      for (int ns = 0; ns < 4; ++ns)
        acc[ms][ns] = __builtin_amdgcn_wmma_f32_16x16x32_bf16(
            false, afrag[ms], false, bfrag[ns], (short)0, acc[ms][ns],
            false, false);

    wait_asynccnt0();   // next tile's DMA has had a full K-step to complete
    __syncthreads();    // publish buffer cur^1 to all waves
  }

  float sc = 1.0f;
  if (MODE == MODE_FINAL || MODE == MODE_RESID) sc = scale[0];

#pragma unroll
  for (int ms = 0; ms < 4; ++ms) {
#pragma unroll
    for (int ns = 0; ns < 4; ++ns) {
#pragma unroll
      for (int r = 0; r < 8; ++r) {
        const int row = m0 + wr * 64 + ms * 16 + r + 8 * h;
        const int col = n0 + wc * 64 + ns * 16 + lo;
        const long idx = ((long)g * M + row) * (long)Nd + col;
        const float v = acc[ms][ns][r];
        if (MODE == MODE_BF16) {
          outb[idx] = f32_to_bf16(v);
        } else if (MODE == MODE_RELU_BF16) {
          outb[idx] = f32_to_bf16(v > 0.f ? v : 0.f);
        } else if (MODE == MODE_F32) {
          outf[idx] = v;
        } else if (MODE == MODE_RESID) {
          const float t = sc * v + res[idx];
          outf[idx] = t;
          outb[idx] = f32_to_bf16(t);
        } else {  // MODE_FINAL
          outf[idx] = sc * v + res[idx];
        }
      }
    }
  }
}

// ---------------------------------------------------------------------------
// Row softmax over last axis: at[row, i] = softmax_i(sc[row, i]), bf16 out.
// One 256-thread block per row, fully coalesced, block reductions in LDS.
// ---------------------------------------------------------------------------
__global__ __launch_bounds__(256)
void softmax_row_kernel(const float* __restrict__ sc,
                        __bf16* __restrict__ at, int L) {
  __shared__ float red[256];
  const long row = blockIdx.x;
  const float* base = sc + row * (long)L;
  const int tid = threadIdx.x;

  float mx = -3.4e38f;
  for (int i = tid; i < L; i += 256) mx = fmaxf(mx, base[i]);
  red[tid] = mx;
  __syncthreads();
  for (int s = 128; s > 0; s >>= 1) {
    if (tid < s) red[tid] = fmaxf(red[tid], red[tid + s]);
    __syncthreads();
  }
  mx = red[0];
  __syncthreads();

  float sum = 0.f;
  for (int i = tid; i < L; i += 256) sum += __expf(base[i] - mx);
  red[tid] = sum;
  __syncthreads();
  for (int s = 128; s > 0; s >>= 1) {
    if (tid < s) red[tid] += red[tid + s];
    __syncthreads();
  }
  const float rs = 1.0f / red[0];

  __bf16* ob = at + row * (long)L;
  for (int i = tid; i < L; i += 256)
    ob[i] = f32_to_bf16(__expf(base[i] - mx) * rs);
}

// ---------------------------------------------------------------------------
// Host-side orchestration
// ---------------------------------------------------------------------------
extern "C" void kernel_launch(void* const* d_in, const int* in_sizes, int n_in,
                              void* d_out, int out_size, void* d_ws, size_t ws_size,
                              hipStream_t stream) {
  (void)in_sizes; (void)n_in; (void)out_size; (void)ws_size;
  const float* x     = (const float*)d_in[0];
  const float* Wq    = (const float*)d_in[1];
  const float* Wk    = (const float*)d_in[2];
  const float* Wv    = (const float*)d_in[3];
  const float* W1    = (const float*)d_in[4];
  const float* W2    = (const float*)d_in[5];
  const float* gamma = (const float*)d_in[6];
  const float* betap = (const float*)d_in[7];
  float* out = (float*)d_out;

  const long NTOK = (long)BB * NNTOK;   // 16384
  const long XE   = NTOK * CCH;         // 8.39M
  const long WE   = (long)CCH * CCH;
  const long SE   = (long)BB * NNTOK * NNTOK;

  char* ws = (char*)d_ws;
  size_t off = 0;
  auto take = [&](size_t bytes) {
    char* p = ws + off; off += (bytes + 255) & ~(size_t)255; return p;
  };

  __bf16* xb  = (__bf16*)take(XE * 2);
  __bf16* wqb = (__bf16*)take(WE * 2);
  __bf16* wkb = (__bf16*)take(WE * 2);
  __bf16* wvb = (__bf16*)take(WE * 2);
  __bf16* w1b = (__bf16*)take(WE * 2);
  __bf16* w2b = (__bf16*)take(WE * 2);
  __bf16* qb  = (__bf16*)take(XE * 2);   // [B,N,C]
  __bf16* kbm = (__bf16*)take(XE * 2);   // [B,N,C]
  __bf16* vtb = (__bf16*)take(XE * 2);   // [B,C,N]  (v transposed)
  float*  sct = (float*) take(SE * 4);   // [B,N(j),N(i)] = scores^T
  __bf16* at  = (__bf16*)take(SE * 2);   // attn^T  [B,j,i]
  float*  of  = (float*) take(XE * 4);   // o (f32 residual)
  __bf16* ob  = (__bf16*)take(XE * 2);   // o (bf16)
  __bf16* hb  = (__bf16*)take(XE * 2);   // relu hidden

  // 1) bf16 conversions
  cvt_f32_bf16_kernel<<<8192, 256, 0, stream>>>(x,  xb,  XE);
  cvt_f32_bf16_kernel<<<1024, 256, 0, stream>>>(Wq, wqb, WE);
  cvt_f32_bf16_kernel<<<1024, 256, 0, stream>>>(Wk, wkb, WE);
  cvt_f32_bf16_kernel<<<1024, 256, 0, stream>>>(Wv, wvb, WE);
  cvt_f32_bf16_kernel<<<1024, 256, 0, stream>>>(W1, w1b, WE);
  cvt_f32_bf16_kernel<<<1024, 256, 0, stream>>>(W2, w2b, WE);

  // 2) q,k projections: out[m,d] = x[m,:].W[d,:]
  dim3 gproj(CCH / 128, NTOK / 256, 1);            // (4,64,1)
  wmma_gemm_tiled<MODE_BF16><<<gproj, 256, 0, stream>>>(
      xb, 0, wqb, 0, (int)NTOK, CCH, CCH, qb, nullptr, nullptr, nullptr);
  wmma_gemm_tiled<MODE_BF16><<<gproj, 256, 0, stream>>>(
      xb, 0, wkb, 0, (int)NTOK, CCH, CCH, kbm, nullptr, nullptr, nullptr);

  // 3) vT[b,c,i] = Wv[c,:].x[b,i,:]  (A shared across batch, sAg=0)
  dim3 gvt(NNTOK / 128, CCH / 256, BB);            // (16,2,8)
  wmma_gemm_tiled<MODE_BF16><<<gvt, 256, 0, stream>>>(
      wvb, 0, xb, (long)NNTOK * CCH, CCH, NNTOK, CCH,
      vtb, nullptr, nullptr, nullptr);

  // 4) scT[b,j,i] = k[b,j,:].q[b,i,:]  (= scores[b,i,j])
  dim3 gsc(NNTOK / 128, NNTOK / 256, BB);          // (16,8,8)
  wmma_gemm_tiled<MODE_F32><<<gsc, 256, 0, stream>>>(
      kbm, (long)NNTOK * CCH, qb, (long)NNTOK * CCH, NNTOK, NNTOK, CCH,
      nullptr, sct, nullptr, nullptr);

  // 5) softmax over i (the contiguous row axis)
  softmax_row_kernel<<<(int)(BB * NNTOK), 256, 0, stream>>>(sct, at, NNTOK);

  // 6) o[b,j,c] = gamma * sum_i attnT[b,j,i]*vT[b,c,i] + x[b,j,c]
  dim3 go(CCH / 128, NNTOK / 256, BB);             // (4,8,8)
  wmma_gemm_tiled<MODE_RESID><<<go, 256, 0, stream>>>(
      at, (long)NNTOK * NNTOK, vtb, (long)CCH * NNTOK, NNTOK, CCH, NNTOK,
      ob, of, x, gamma);

  // 7) FFN: h = relu(o @ W1^T); out = beta*(h @ W2^T) + o
  wmma_gemm_tiled<MODE_RELU_BF16><<<gproj, 256, 0, stream>>>(
      ob, 0, w1b, 0, (int)NTOK, CCH, CCH, hb, nullptr, nullptr, nullptr);
  wmma_gemm_tiled<MODE_FINAL><<<gproj, 256, 0, stream>>>(
      hb, 0, w2b, 0, (int)NTOK, CCH, CCH, nullptr, out, of, betap);
}
